// Generator_25941602468305
// MI455X (gfx1250) — matrix operand order repaired and checked by reading, not compiled, not run on hardware
//
#include <hip/hip_runtime.h>
#include <hip/hip_bf16.h>

typedef __bf16 bf16_t;
typedef __attribute__((ext_vector_type(16))) __bf16 v16bf;
typedef __attribute__((ext_vector_type(8)))  __bf16 v8bf;
typedef __attribute__((ext_vector_type(8)))  float  v8f;

#define WMMA_BF16(a, b, c) \
  __builtin_amdgcn_wmma_f32_16x16x32_bf16(false, (a), false, (b), (short)0, (c), false, false)

__device__ __forceinline__ float sigf(float x) { return 1.0f / (1.0f + __expf(-x)); }

// A-fragment: 16x32 bf16 tile at (row0, k0); lanes 0-15 rows M=0..15 with K 0..7 & 16..23,
// lanes 16-31 same rows with K 8..15 & 24..31 (gfx1250 16-bit A layout).
__device__ __forceinline__ v16bf load_a_frag(const bf16_t* tile, int ld, int lane) {
  const bf16_t* p = tile + (size_t)(lane & 15) * ld + ((lane >> 4) << 3);
  v8bf lo = *(const v8bf*)(p);
  v8bf hi = *(const v8bf*)(p + 16);
  v16bf r;
#pragma unroll
  for (int i = 0; i < 8; ++i) { r[i] = lo[i]; r[i + 8] = hi[i]; }
  return r;
}

// B-fragment: B = W^T tile; W row-major [N,K]; lane (0-15) -> column n (= W row), holds K 0..7 & 16..23;
// lanes 16-31 hold K 8..15 & 24..31.
__device__ __forceinline__ v16bf load_b_frag(const bf16_t* wtile, int ld, int lane) {
  const bf16_t* p = wtile + (size_t)(lane & 15) * ld + ((lane >> 4) << 3);
  v8bf lo = *(const v8bf*)(p);
  v8bf hi = *(const v8bf*)(p + 16);
  v16bf r;
#pragma unroll
  for (int i = 0; i < 8; ++i) { r[i] = lo[i]; r[i + 8] = hi[i]; }
  return r;
}

// ---------------------------------------------------------------------------
// Async global->LDS staging of a 128x32 bf16 A-tile (CDNA5 ASYNCcnt path).
// 512 chunks of 16B; 256 threads issue 2 global_load_async_to_lds_b128 each.
// LDS byte address = low 32 bits of the flat pointer (aperture mapping).
// ---------------------------------------------------------------------------
#define ATILE_STR 40  // bf16 per staged row (32 + 8 pad) -> 16 rows hit 16 distinct banks

__device__ __forceinline__ void stage_a_tile(const bf16_t* __restrict__ A, bf16_t* dstbase,
                                             int mblk, int K, int kt, int tid) {
#pragma unroll
  for (int c2 = 0; c2 < 2; ++c2) {
    int linear = c2 * 256 + tid;          // 4 consecutive tids cover one 32-elem row
    int row  = linear >> 2;
    int kcol = (linear & 3) << 3;
    const bf16_t* g = A + (size_t)(mblk + row) * K + kt * 32 + kcol;
    unsigned l = (unsigned)(unsigned long long)(dstbase + row * ATILE_STR + kcol);
    asm volatile("global_load_async_to_lds_b128 %0, %1, off" :: "v"(l), "v"(g) : "memory");
  }
}

// ---------------------------------------------------------------------------
// WMMA GEMM: C[M,N] = A[M,K](bf16) @ W[N,K](bf16)^T + bias[N] (+ z*wz rank-1 if HASZ)
// Block 256 threads = 8 waves; block tile 128x128; wave tile 64x32 (4x2 wmma tiles).
// A tile double-buffered in LDS via async loads; B streamed from global (L2-resident weights).
// ---------------------------------------------------------------------------
template <bool HASZ>
__global__ void __launch_bounds__(256) wmma_gemm_bf16(
    const bf16_t* __restrict__ A, const bf16_t* __restrict__ W,
    const float* __restrict__ bias, const float* __restrict__ zvec,
    const float* __restrict__ wz, float* __restrict__ C,
    int M, int N, int K, int zT) {
  __shared__ __align__(16) bf16_t asmem[2][128 * ATILE_STR];  // 20 KB
  const int tid  = threadIdx.x;
  const int lane = tid & 31;
  const int wid  = tid >> 5;
  const int mblk = blockIdx.y * 128;
  const int m0w  = (wid & 1) * 64;
  const int n0   = blockIdx.x * 128 + (wid >> 1) * 32;

  v8f acc[4][2];
#pragma unroll
  for (int mt = 0; mt < 4; ++mt)
#pragma unroll
    for (int j = 0; j < 2; ++j)
#pragma unroll
      for (int r = 0; r < 8; ++r) acc[mt][j][r] = 0.0f;

  const int KT = K >> 5;
  stage_a_tile(A, asmem[0], mblk, K, 0, tid);
  for (int kt = 0; kt < KT; ++kt) {
    if (kt + 1 < KT) {
      stage_a_tile(A, asmem[(kt + 1) & 1], mblk, K, kt + 1, tid);
      asm volatile("s_wait_asynccnt 0x2" ::: "memory");  // current tile's loads done
    } else {
      asm volatile("s_wait_asynccnt 0x0" ::: "memory");
    }
    __syncthreads();

    const bf16_t* at = asmem[kt & 1];
    const int k0 = kt * 32;
    v16bf a[4];
#pragma unroll
    for (int mt = 0; mt < 4; ++mt)
      a[mt] = load_a_frag(at + (size_t)(m0w + mt * 16) * ATILE_STR, ATILE_STR, lane);
#pragma unroll
    for (int j = 0; j < 2; ++j) {
      v16bf b = load_b_frag(W + (size_t)(n0 + j * 16) * K + k0, K, lane);
#pragma unroll
      for (int mt = 0; mt < 4; ++mt) acc[mt][j] = WMMA_BF16(a[mt], b, acc[mt][j]);
    }
    __syncthreads();  // all waves done reading this buffer before it is restaged
  }

  // Branch-free epilogue: per-column terms hoisted out of the store loops.
  float bj[2], wzj[2];
#pragma unroll
  for (int j = 0; j < 2; ++j) {
    int col = n0 + j * 16 + (lane & 15);
    bj[j]  = bias[col];
    wzj[j] = HASZ ? wz[col] : 0.0f;
  }
#pragma unroll
  for (int mt = 0; mt < 4; ++mt)
#pragma unroll
    for (int r = 0; r < 8; ++r) {
      int row = mblk + m0w + mt * 16 + r + ((lane >> 4) << 3);
      float zr = HASZ ? zvec[(size_t)(row & 63) * zT + (row >> 6)] : 0.0f;
#pragma unroll
      for (int j = 0; j < 2; ++j) {
        int col = n0 + j * 16 + (lane & 15);
        float v = acc[mt][j][r] + bj[j];
        if (HASZ) v += zr * wzj[j];
        C[(size_t)row * N + col] = v;
      }
    }
}

// ---------------------------------------------------------------------------
// Persistent bidirectional LSTM scan, H=256, B=64, 4H=1024. grid.x = 2 (dir).
// 1024 threads = 32 waves; per step each wave does 2 N-tiles x 4 M-tiles x 8 K-steps of WMMA.
// Gates (64x1024 fp32) + h (64x256 bf16, padded) live in 290KB dynamic LDS.
// ---------------------------------------------------------------------------
#define HSTR 264   // bf16 elems per h row (256 + 8 pad), 528B (16B multiple)
#define GSTR 1028  // fp32 elems per gates row (1024 + 4 pad)

__global__ void __launch_bounds__(1024) lstm_scan_big(
    const float* __restrict__ xg0, const float* __restrict__ xg1,
    const bf16_t* __restrict__ whh_base, size_t whh_dir_stride,
    bf16_t* __restrict__ outb, float* __restrict__ outf, int T) {
  extern __shared__ char smem[];
  float*  gates = (float*)smem;                            // 64*GSTR fp32
  bf16_t* hbuf  = (bf16_t*)(smem + (size_t)64 * GSTR * 4); // 64*HSTR bf16

  const int tid = threadIdx.x, lane = tid & 31, wid = tid >> 5;
  const int dir = blockIdx.x;
  const float*  xg  = dir ? xg1 : xg0;
  const bf16_t* Whh = whh_base + (size_t)dir * whh_dir_stride;
  const int coloff = dir * 256;

  for (int i = tid; i < 64 * HSTR; i += 1024) hbuf[i] = (bf16_t)0.0f;
  float c[16];
#pragma unroll
  for (int u = 0; u < 16; ++u) c[u] = 0.0f;
  const int eb = tid >> 4, ej0 = (tid & 15) << 4;
  const int nt0 = wid * 2;
  __syncthreads();

  for (int it = 0; it < T; ++it) {
    const int t = dir ? (T - 1 - it) : it;

    // --- phase 1: gates = h_{t-1} @ Whh^T + xg[t]
    v8f acc[4][2];
#pragma unroll
    for (int mt = 0; mt < 4; ++mt)
#pragma unroll
      for (int j = 0; j < 2; ++j)
#pragma unroll
        for (int r = 0; r < 8; ++r) acc[mt][j][r] = 0.0f;

#pragma unroll 2
    for (int ks = 0; ks < 8; ++ks) {
      const int k0 = ks * 32;
      if (ks < 7)  // WGP-scope prefetch of the next Whh k-slice (locality 3 -> near cache)
        __builtin_prefetch(Whh + (size_t)(nt0 * 16) * 256 + k0 + 32, 0, 3);
      v16bf a[4];
#pragma unroll
      for (int mt = 0; mt < 4; ++mt)
        a[mt] = load_a_frag(hbuf + (size_t)(mt * 16) * HSTR + k0, HSTR, lane);
#pragma unroll
      for (int j = 0; j < 2; ++j) {
        v16bf b = load_b_frag(Whh + (size_t)((nt0 + j) * 16) * 256 + k0, 256, lane);
#pragma unroll
        for (int mt = 0; mt < 4; ++mt) acc[mt][j] = WMMA_BF16(a[mt], b, acc[mt][j]);
      }
    }
    const float* xgt = xg + (size_t)t * 64 * 1024;
#pragma unroll
    for (int mt = 0; mt < 4; ++mt)
#pragma unroll
      for (int j = 0; j < 2; ++j)
#pragma unroll
        for (int r = 0; r < 8; ++r) {
          int row = mt * 16 + r + ((lane >> 4) << 3);
          int col = (nt0 + j) * 16 + (lane & 15);
          gates[row * GSTR + col] = acc[mt][j][r] + xgt[(size_t)row * 1024 + col];
        }
    __syncthreads();

    // --- phase 2: elementwise LSTM cell; thread owns (b=eb, units ej0..ej0+15)
    {
      const float* g0 = gates + eb * GSTR;
      bf16_t* hrow = hbuf + eb * HSTR + ej0;
      size_t obase = ((size_t)t * 64 + eb) * 512 + coloff + ej0;
#pragma unroll
      for (int u = 0; u < 16; ++u) {
        float gi = g0[ej0 + u];
        float gf = g0[256 + ej0 + u];
        float gg = g0[512 + ej0 + u];
        float go = g0[768 + ej0 + u];
        float ci = sigf(gf) * c[u] + sigf(gi) * tanhf(gg);
        c[u] = ci;
        float h = sigf(go) * tanhf(ci);
        hrow[u] = (bf16_t)h;
        outb[obase + u] = (bf16_t)h;
        if (outf) outf[obase + u] = h;
      }
    }
    __syncthreads();
  }
}

// ---------------------------------------------------------------------------
// Small LSTM scan (S=64) fused with logit / log-likelihood reduction.
// 512 threads = 16 waves; per step each wave does 1 N-tile x 4 M-tiles x 2 K-steps.
// Logit at step t uses h_{t-1} (evaluated before the state update) == s_prev.
// ---------------------------------------------------------------------------
#define SGSTR 260  // gates fp32 row stride (256 + 4)
#define SFSTR 68   // h fp32 row stride (64 + 4)
#define SHSTR 72   // h bf16 row stride (64 + 8), 144B (16B multiple)

__global__ void __launch_bounds__(512) lstm_scan_small(
    const float* __restrict__ xg, const bf16_t* __restrict__ whh,
    const float* __restrict__ baseLogit, const float* __restrict__ z,
    const float* __restrict__ WoS, float* __restrict__ outLogp, int T) {
  extern __shared__ char smem[];
  float*  gates = (float*)smem;               // 64*SGSTR
  float*  hf    = gates + 64 * SGSTR;         // 64*SFSTR fp32 h
  bf16_t* hb    = (bf16_t*)(hf + 64 * SFSTR); // 64*SHSTR bf16 h

  const int tid = threadIdx.x, lane = tid & 31, wid = tid >> 5;
  for (int i = tid; i < 64 * SFSTR; i += 512) hf[i] = 0.0f;
  for (int i = tid; i < 64 * SHSTR; i += 512) hb[i] = (bf16_t)0.0f;
  float c[8];
#pragma unroll
  for (int u = 0; u < 8; ++u) c[u] = 0.0f;
  float logp = 0.0f;
  const int eb = tid >> 3, ej0 = (tid & 7) << 3;
  __syncthreads();

  for (int t = 0; t < T; ++t) {
    // --- gates = h_{t-1} @ Whh_s^T + xg_s[t]
    v8f acc[4];
#pragma unroll
    for (int mt = 0; mt < 4; ++mt)
#pragma unroll
      for (int r = 0; r < 8; ++r) acc[mt][r] = 0.0f;
#pragma unroll
    for (int ks = 0; ks < 2; ++ks) {
      const int k0 = ks * 32;
      v16bf b = load_b_frag(whh + (size_t)(wid * 16) * 64 + k0, 64, lane);
#pragma unroll
      for (int mt = 0; mt < 4; ++mt) {
        v16bf a = load_a_frag(hb + (size_t)(mt * 16) * SHSTR + k0, SHSTR, lane);
        acc[mt] = WMMA_BF16(a, b, acc[mt]);
      }
    }
    const float* xgt = xg + (size_t)t * 64 * 256;
#pragma unroll
    for (int mt = 0; mt < 4; ++mt)
#pragma unroll
      for (int r = 0; r < 8; ++r) {
        int row = mt * 16 + r + ((lane >> 4) << 3);
        int col = wid * 16 + (lane & 15);
        gates[row * SGSTR + col] = acc[mt][r] + xgt[(size_t)row * 256 + col];
      }

    // --- logit with s_prev = h_{t-1}; accumulate log-likelihood (threads 0..63 own one batch row)
    if (tid < 64) {
      float a = baseLogit[(size_t)t * 64 + tid];
      const float* hr = hf + tid * SFSTR;
#pragma unroll 8
      for (int k = 0; k < 64; ++k) a += hr[k] * WoS[k];
      float p  = sigf(a);
      float zb = z[(size_t)tid * T + t];
      float lik = zb * p + (1.0f - zb) * (1.0f - p);
      logp += __logf(fmaxf(lik, 1e-30f));
    }
    __syncthreads();

    // --- elementwise cell update; thread owns (b=eb, units ej0..ej0+7)
    {
      float* g0 = gates + eb * SGSTR;
#pragma unroll
      for (int u = 0; u < 8; ++u) {
        int j = ej0 + u;
        float gi = g0[j], gf = g0[64 + j], gg = g0[128 + j], go = g0[192 + j];
        float ci = sigf(gf) * c[u] + sigf(gi) * tanhf(gg);
        c[u] = ci;
        float h = sigf(go) * tanhf(ci);
        hf[eb * SFSTR + j] = h;
        hb[eb * SHSTR + j] = (bf16_t)h;
      }
    }
    __syncthreads();
  }
  if (tid < 64) outLogp[tid] = logp;
}

// ---------------------------------------------------------------------------
// Helper kernels
// ---------------------------------------------------------------------------
__global__ void f2bf_kernel(bf16_t* dst, const float* __restrict__ src,
                            int rows, int cols, int ld) {
  int i = blockIdx.x * 256 + threadIdx.x;
  if (i >= rows * cols) return;
  int r = i / cols, cc = i - r * cols;
  dst[i] = (bf16_t)src[(size_t)r * ld + cc];
}

__global__ void vec_add_kernel(float* dst, const float* __restrict__ a,
                               const float* __restrict__ b, int n) {
  int i = blockIdx.x * 256 + threadIdx.x;
  if (i < n) dst[i] = a[i] + b[i];
}

__global__ void col_extract_kernel(float* dst, const float* __restrict__ src,
                                   int n, int ld, int col) {
  int i = blockIdx.x * 256 + threadIdx.x;
  if (i < n) dst[i] = src[(size_t)i * ld + col];
}

__global__ void embed_kernel(const int* __restrict__ idx, const float* __restrict__ E,
                             bf16_t* __restrict__ xbf, int B, int T, int D) {
  int m = blockIdx.x;        // m = t*B + b
  int d = threadIdx.x;
  int b = m % B, t = m / B;
  int row = idx[b * T + t];
  xbf[(size_t)m * D + d] = (bf16_t)E[(size_t)row * D + d];
}

__global__ void base_logit_kernel(const float* __restrict__ hid, const float* __restrict__ Wo,
                                  const float* __restrict__ bo, float* __restrict__ out,
                                  int M, int K) {
  int m = blockIdx.x * 256 + threadIdx.x;
  if (m >= M) return;
  const float* r = hid + (size_t)m * K;
  float acc = bo[0];
#pragma unroll 8
  for (int k = 0; k < K; ++k) acc += r[k] * Wo[k];
  out[m] = acc;
}

// ---------------------------------------------------------------------------
extern "C" void kernel_launch(void* const* d_in, const int* in_sizes, int n_in,
                              void* d_out, int out_size, void* d_ws, size_t ws_size,
                              hipStream_t stream) {
  (void)in_sizes; (void)n_in; (void)out_size; (void)ws_size;
  const int B = 64, T = 1024, D = 256, H = 256, S = 64;
  const int G  = 4 * H;   // 1024
  const int GS = 4 * S;   // 256
  const int M  = T * B;   // 65536

  const int*   x_index = (const int*)  d_in[0];
  const float* z       = (const float*)d_in[1];
  const float* embed_W = (const float*)d_in[2];
  const float* W_ih_l0 = (const float*)d_in[3];
  const float* W_hh_l0 = (const float*)d_in[4];
  const float* b_ih_l0 = (const float*)d_in[5];
  const float* b_hh_l0 = (const float*)d_in[6];
  const float* W_ih_l1 = (const float*)d_in[7];
  const float* W_hh_l1 = (const float*)d_in[8];
  const float* b_ih_l1 = (const float*)d_in[9];
  const float* b_hh_l1 = (const float*)d_in[10];
  const float* W_ih_s  = (const float*)d_in[11];
  const float* W_hh_s  = (const float*)d_in[12];
  const float* b_ih_s  = (const float*)d_in[13];
  const float* b_hh_s  = (const float*)d_in[14];
  const float* W_o     = (const float*)d_in[15];
  const float* b_o     = (const float*)d_in[16];

  char* ws = (char*)d_ws;
  size_t off = 0;
  auto alloc = [&](size_t bytes) -> void* {
    void* p = ws + off;
    off = (off + bytes + 255) & ~(size_t)255;
    return p;
  };

  bf16_t* xbf    = (bf16_t*)alloc((size_t)M * D * 2);
  bf16_t* wihl0b = (bf16_t*)alloc((size_t)2 * G * D * 2);
  bf16_t* whhl0b = (bf16_t*)alloc((size_t)2 * G * H * 2);
  bf16_t* wihl1b = (bf16_t*)alloc((size_t)2 * G * (2 * H) * 2);
  bf16_t* whhl1b = (bf16_t*)alloc((size_t)2 * G * H * 2);
  bf16_t* wihsb  = (bf16_t*)alloc((size_t)GS * (2 * H) * 2);
  bf16_t* whhsb  = (bf16_t*)alloc((size_t)GS * S * 2);
  float*  wz     = (float*) alloc((size_t)GS * 4);
  float*  bs0    = (float*) alloc((size_t)2 * G * 4);
  float*  bs1    = (float*) alloc((size_t)2 * G * 4);
  float*  bss    = (float*) alloc((size_t)GS * 4);
  float*  xgA    = (float*) alloc((size_t)M * G * 4);
  float*  xgB    = (float*) alloc((size_t)M * G * 4);
  bf16_t* out0b  = (bf16_t*)alloc((size_t)M * (2 * H) * 2);
  bf16_t* out1b  = (bf16_t*)alloc((size_t)M * (2 * H) * 2);
  float*  out1f  = (float*) alloc((size_t)M * (2 * H) * 4);
  float*  basel  = (float*) alloc((size_t)M * 4);

  // weight conversion / bias folding
  f2bf_kernel<<<(2 * G * D + 255) / 256, 256, 0, stream>>>(wihl0b, W_ih_l0, 2 * G, D, D);
  f2bf_kernel<<<(2 * G * H + 255) / 256, 256, 0, stream>>>(whhl0b, W_hh_l0, 2 * G, H, H);
  f2bf_kernel<<<(2 * G * 2 * H + 255) / 256, 256, 0, stream>>>(wihl1b, W_ih_l1, 2 * G, 2 * H, 2 * H);
  f2bf_kernel<<<(2 * G * H + 255) / 256, 256, 0, stream>>>(whhl1b, W_hh_l1, 2 * G, H, H);
  f2bf_kernel<<<(GS * 2 * H + 255) / 256, 256, 0, stream>>>(wihsb, W_ih_s, GS, 2 * H, 2 * H + 1);
  f2bf_kernel<<<(GS * S + 255) / 256, 256, 0, stream>>>(whhsb, W_hh_s, GS, S, S);
  col_extract_kernel<<<1, 256, 0, stream>>>(wz, W_ih_s, GS, 2 * H + 1, 2 * H);
  vec_add_kernel<<<(2 * G + 255) / 256, 256, 0, stream>>>(bs0, b_ih_l0, b_hh_l0, 2 * G);
  vec_add_kernel<<<(2 * G + 255) / 256, 256, 0, stream>>>(bs1, b_ih_l1, b_hh_l1, 2 * G);
  vec_add_kernel<<<1, 256, 0, stream>>>(bss, b_ih_s, b_hh_s, GS);

  embed_kernel<<<M, D, 0, stream>>>(x_index, embed_W, xbf, B, T, D);

  const int SMEM_BIG   = 64 * GSTR * 4 + 64 * HSTR * 2;                    // ~290 KB
  const int SMEM_SMALL = 64 * SGSTR * 4 + 64 * SFSTR * 4 + 64 * SHSTR * 2; // ~91 KB
  hipFuncSetAttribute((const void*)lstm_scan_big,
                      hipFuncAttributeMaxDynamicSharedMemorySize, SMEM_BIG);
  hipFuncSetAttribute((const void*)lstm_scan_small,
                      hipFuncAttributeMaxDynamicSharedMemorySize, SMEM_SMALL);

  dim3 gemmBlk(256);
  // layer 0: input GEMMs (fwd/bwd) then bidirectional scan
  wmma_gemm_bf16<false><<<dim3(G / 128, M / 128), gemmBlk, 0, stream>>>(
      xbf, wihl0b, bs0, nullptr, nullptr, xgA, M, G, D, T);
  wmma_gemm_bf16<false><<<dim3(G / 128, M / 128), gemmBlk, 0, stream>>>(
      xbf, wihl0b + (size_t)G * D, bs0 + G, nullptr, nullptr, xgB, M, G, D, T);
  lstm_scan_big<<<2, 1024, SMEM_BIG, stream>>>(xgA, xgB, whhl0b, (size_t)G * H,
                                               out0b, nullptr, T);
  // layer 1
  wmma_gemm_bf16<false><<<dim3(G / 128, M / 128), gemmBlk, 0, stream>>>(
      out0b, wihl1b, bs1, nullptr, nullptr, xgA, M, G, 2 * H, T);
  wmma_gemm_bf16<false><<<dim3(G / 128, M / 128), gemmBlk, 0, stream>>>(
      out0b, wihl1b + (size_t)G * 2 * H, bs1 + G, nullptr, nullptr, xgB, M, G, 2 * H, T);
  lstm_scan_big<<<2, 1024, SMEM_BIG, stream>>>(xgA, xgB, whhl1b, (size_t)G * H,
                                               out1b, out1f, T);
  // small LSTM input GEMM: hidden part (K=512) + rank-1 z term in epilogue (reuses xgA)
  wmma_gemm_bf16<true><<<dim3(GS / 128, M / 128), gemmBlk, 0, stream>>>(
      out1b, wihsb, bss, z, wz, xgA, M, GS, 2 * H, T);
  // parallel part of the logit: hidden(512) . Wo[:512] + b_o
  base_logit_kernel<<<(M + 255) / 256, 256, 0, stream>>>(out1f, W_o, b_o, basel, M, 2 * H);
  // small scan + fused logit / log-likelihood reduction -> d_out[64]
  lstm_scan_small<<<1, 512, SMEM_SMALL, stream>>>(xgA, whhsb, basel, z, W_o + 2 * H,
                                                  (float*)d_out, T);
}
